// TextureBranch_57904749085060
// MI455X (gfx1250) — matrix operand order repaired and checked by reading, not compiled, not run on hardware
//
#include <hip/hip_runtime.h>
#include <hip/hip_bf16.h>
#include <math.h>

typedef __attribute__((ext_vector_type(16))) _Float16 v16h;
typedef __attribute__((ext_vector_type(8)))  _Float16 v8h;
typedef __attribute__((ext_vector_type(8)))  float    v8f;

#define BN_EPS 1e-5f

// ---------------------------------------------------------------------------
// Kernel 0: v8h zero-fill (halo init; re-run every call, no persistent state).
// ---------------------------------------------------------------------------
__global__ __launch_bounds__(256)
void fill_zero_kernel(_Float16* __restrict__ p, int nvec) {  // nvec = f16count/8
  const int i = blockIdx.x * 256 + threadIdx.x;
  if (i < nvec) ((v8h*)p)[i] = (v8h){};
}

// ---------------------------------------------------------------------------
// Kernel 1: per-64x64-tile sum / sumsq (reads the 100MB input exactly once).
// Patches (128x128, stride 64) are 2x2 unions of these tiles.
// grid = B*3*64 blocks of 256 threads.
// ---------------------------------------------------------------------------
__global__ __launch_bounds__(256)
void tile_stats_kernel(const float* __restrict__ x,
                       float* __restrict__ tsum, float* __restrict__ tsq) {
  const int tile = blockIdx.x;            // b*192 + c*64 + ty*8 + tx
  const int tx = tile & 7;
  const int ty = (tile >> 3) & 7;
  const int c  = (tile >> 6) % 3;
  const int b  = tile / 192;
  const float* base =
      x + (((size_t)b * 3 + c) * 512 + (size_t)ty * 64) * 512 + (size_t)tx * 64;
  float s = 0.f, q = 0.f;
  for (int i = threadIdx.x; i < 64 * 64; i += 256) {
    const int r = i >> 6, col = i & 63;
    const float v = base[(size_t)r * 512 + col];
    s += v;
    q += v * v;
  }
  __shared__ float ss[256];
  __shared__ float sq[256];
  ss[threadIdx.x] = s;
  sq[threadIdx.x] = q;
  __syncthreads();
  for (int st = 128; st > 0; st >>= 1) {
    if ((int)threadIdx.x < st) {
      ss[threadIdx.x] += ss[threadIdx.x + st];
      sq[threadIdx.x] += sq[threadIdx.x + st];
    }
    __syncthreads();
  }
  if (threadIdx.x == 0) {
    tsum[tile] = ss[0];
    tsq[tile]  = sq[0];
  }
}

// ---------------------------------------------------------------------------
// Kernel 2: 49 patch variances (ddof=1) -> argmin per batch. grid=B, block=64.
// ---------------------------------------------------------------------------
__global__ __launch_bounds__(64)
void select_patch_kernel(const float* __restrict__ tsum,
                         const float* __restrict__ tsq,
                         int* __restrict__ idx) {
  const int b = blockIdx.x;
  const int p = threadIdx.x;
  float var = 3.4e38f;
  if (p < 49) {
    const int pi = p / 7, pj = p % 7;
    float s = 0.f, q = 0.f;
    for (int c = 0; c < 3; ++c)
      for (int di = 0; di < 2; ++di)
        for (int dj = 0; dj < 2; ++dj) {
          const int t = (b * 3 + c) * 64 + (pi + di) * 8 + (pj + dj);
          s += tsum[t];
          q += tsq[t];
        }
    const float n = 3.0f * 128.0f * 128.0f;
    var = (q - s * s / n) / (n - 1.0f);
  }
  __shared__ float sv[64];
  __shared__ int   si[64];
  sv[p] = var;
  si[p] = p;
  __syncthreads();
  for (int st = 32; st > 0; st >>= 1) {
    if (p < st) {
      const float ov = sv[p + st];
      const int   oi = si[p + st];
      if (ov < sv[p] || (ov == sv[p] && oi < si[p])) { sv[p] = ov; si[p] = oi; }
    }
    __syncthreads();
  }
  if (p == 0) idx[b] = si[0];
}

// ---------------------------------------------------------------------------
// Kernel 3: gather selected patch -> padded NHWC f16, 8 channels (3 real,
// 5 zero). sel dims per image: 131 rows x 130 cols x 8 (1 top halo, 2 bottom
// halo rows for the K-padding taps, 1 left/right). One b128 store per pixel.
// grid = B*128 rows, block = 128.
// ---------------------------------------------------------------------------
__global__ __launch_bounds__(128)
void gather_patch_kernel(const float* __restrict__ x,
                         const int* __restrict__ idx,
                         _Float16* __restrict__ sel) {
  const int row = blockIdx.x;       // b*128 + y
  const int y = row & 127;
  const int b = row >> 7;
  const int p = idx[b];
  const int pi = p / 7, pj = p % 7;
  const int xc = threadIdx.x;
  const float* src =
      x + ((size_t)b * 3 * 512 + (size_t)(pi * 64 + y)) * 512 + (size_t)(pj * 64) + xc;
  v8h v = {};
  v[0] = (_Float16)src[0];
  v[1] = (_Float16)src[512 * 512];
  v[2] = (_Float16)src[2 * 512 * 512];
  _Float16* dst = sel + (((size_t)b * 131 + (y + 1)) * 130 + (xc + 1)) * 8;
  *(v8h*)dst = v;
}

// ---------------------------------------------------------------------------
// Kernel 4: repack OIHW f32 weights -> [tap][co][ci] f16, zero-padded in both
// ci (CIsrc -> CIp) and tap (9 -> TAPSp). The zero taps make the K-padding
// chunks contribute exactly zero regardless of what A loads there.
// ---------------------------------------------------------------------------
__global__ __launch_bounds__(256)
void repack_weights_kernel(const float* __restrict__ w, _Float16* __restrict__ wpk,
                           int CIsrc, int CIp, int CO, int total) {
  const int i = blockIdx.x * 256 + threadIdx.x;
  if (i >= total) return;
  const int ci = i % CIp;
  const int co = (i / CIp) % CO;
  const int t  = i / (CIp * CO);
  float v = 0.f;
  if (t < 9 && ci < CIsrc)
    v = w[((size_t)co * CIsrc + ci) * 9 + t];     // w[co][ci][ky][kx], t=ky*3+kx
  wpk[i] = (_Float16)v;
}

// ---------------------------------------------------------------------------
// Kernel 5: fused conv3x3(SAME)+bias+BN+ReLU, implicit GEMM on WMMA.
// Input: spatially zero-padded NHWC f16 (PR rows x (W+2) cols x CI), so all
// fragment loads are UNCONDITIONAL b128s. The only runtime input to a
// fragment offset is half = lane>>4, so both half-variants of every offset
// are compile-time constants and each address is one select (becomes
// base + 8*half when both variants share a tap, i.e. all CI>=32 offsets).
//   A 16x32: lane L -> M=L%16; runs at K = half*8 + {0..7, 16..23}
//   B 32x16: lane L -> N=L%16; groups at K = half*8 + {0..7, 16..23}
//   C/D    : VGPR r, lanes0-15 -> M=r, lanes16-31 -> M=r+8, N=L%16
// 8 waves/block; each wave: 16 pixels x (16*NT) out-channels, A fragment
// reused across NT WMMAs per chunk.
// ---------------------------------------------------------------------------
template <int CI, int CO, int H, int W, int PR, int NT>
__global__ __launch_bounds__(256)
void conv_bn_relu_wmma_kernel(const _Float16* __restrict__ in,
                              const _Float16* __restrict__ wpk,
                              const float* __restrict__ bias,
                              const float* __restrict__ g,
                              const float* __restrict__ bt,
                              const float* __restrict__ rm,
                              const float* __restrict__ rv,
                              _Float16* __restrict__ out) {
  constexpr int NCH = (CI * 9 + 31) / 32;   // K chunks of 32
  constexpr int MTI = (H * W) / 16;         // M-tiles per image
  constexpr int NG  = CO / (16 * NT);       // N-groups per image
  constexpr int WP  = W + 2;                // padded row width

  const int lane = threadIdx.x & 31;
  const int wave = blockIdx.x * 8 + (threadIdx.x >> 5);
  const int b    = wave / (MTI * NG);
  const int u    = wave % (MTI * NG);
  const int mt   = u % MTI;
  const int ng   = u / MTI;
  const int co0  = ng * (16 * NT);
  const int y    = (mt * 16) / W;
  const int x0   = (mt * 16) % W;
  const int mlane = lane & 15;
  const int half  = lane >> 4;              // only runtime input to K offsets
  const int xa    = x0 + mlane;             // this lane's output x (A rows)

  const _Float16* Abase = in + ((size_t)b * PR * WP + (size_t)y * WP + xa) * CI;

  v8f acc[NT];
#pragma unroll
  for (int nt = 0; nt < NT; ++nt) acc[nt] = (v8f){};

#pragma unroll
  for (int kc = 0; kc < NCH; ++kc) {
    // ---- A fragment: 2 runs, offset = select(half, const, const) ----
    v16h afrag;
#pragma unroll
    for (int r = 0; r < 2; ++r) {
      const int K0 = kc * 32 + r * 16;                 // half = 0 variant
      const int K1 = K0 + 8;                           // half = 1 variant
      const int t0 = K0 / CI, t1 = K1 / CI;
      const int O0 = ((t0 / 3) * WP + (t0 % 3)) * CI + (K0 % CI);
      const int O1 = ((t1 / 3) * WP + (t1 % 3)) * CI + (K1 % CI);
      const int off = half ? O1 : O0;                  // folds if O0 == O1
      const v8h v = *(const v8h*)(Abase + off);
#pragma unroll
      for (int e = 0; e < 8; ++e) afrag[r * 8 + e] = v[e];
    }
    // ---- B fragments + WMMA (A reused NT times) ----
#pragma unroll
    for (int nt = 0; nt < NT; ++nt) {
      const int co = co0 + nt * 16 + mlane;
      const _Float16* Bbase = wpk + (size_t)co * CI;
      v16h bfrag;
#pragma unroll
      for (int gq = 0; gq < 2; ++gq) {
        const int K0 = kc * 32 + gq * 16;              // half = 0 variant
        const int K1 = K0 + 8;                         // half = 1 variant
        const int O0 = (K0 / CI) * (CO * CI) + (K0 % CI);
        const int O1 = (K1 / CI) * (CO * CI) + (K1 % CI);
        const int off = half ? O1 : O0;
        const v8h v = *(const v8h*)(Bbase + off);
#pragma unroll
        for (int e = 0; e < 8; ++e) bfrag[gq * 8 + e] = v[e];
      }
      acc[nt] = __builtin_amdgcn_wmma_f32_16x16x32_f16(
          /*neg_a=*/false, afrag, /*neg_b=*/false, bfrag,
          /*c_mod=*/(short)0, acc[nt], /*reuse_a=*/false, /*reuse_b=*/false);
    }
  }

  // Epilogue: bias + BN + ReLU -> unpadded NHWC f16 (half-wave-contiguous).
#pragma unroll
  for (int nt = 0; nt < NT; ++nt) {
    const int co = co0 + nt * 16 + mlane;
    const float scale = g[co] * rsqrtf(rv[co] + BN_EPS);
    const float shift = bt[co] - rm[co] * scale;
    const float bi    = bias[co];
    _Float16* obase = out + (((size_t)b * H + y) * W + x0 + half * 8) * CO + co;
#pragma unroll
    for (int r = 0; r < 8; ++r) {
      float v = (acc[nt][r] + bi) * scale + shift;
      v = fmaxf(v, 0.f);
      obase[(size_t)r * CO] = (_Float16)v;
    }
  }
}

// ---------------------------------------------------------------------------
// Kernel 6: 2x2 max-pool stride 2, NHWC f16 (unpadded in) -> padded-NHWC out
// interior at (yo+1, xo+1) with row width Wo+2.
// ---------------------------------------------------------------------------
__global__ __launch_bounds__(256)
void maxpool2_nhwc_kernel(const _Float16* __restrict__ in, _Float16* __restrict__ out,
                          int C, int H, int W, int total) {
  const int i = blockIdx.x * 256 + threadIdx.x;
  if (i >= total) return;
  const int Wo = W >> 1, Ho = H >> 1;
  const int c  = i % C;
  const int xo = (i / C) % Wo;
  const int yo = (i / C / Wo) % Ho;
  const int b  = i / (C * Wo * Ho);
  const _Float16* p = in + (((size_t)b * H + (size_t)yo * 2) * W + (size_t)xo * 2) * C + c;
  const size_t rs = (size_t)W * C;
  const float m = fmaxf(fmaxf((float)p[0], (float)p[C]),
                        fmaxf((float)p[rs], (float)p[rs + C]));
  out[(((size_t)b * (Ho + 2) + (yo + 1)) * (Wo + 2) + (xo + 1)) * C + c] = (_Float16)m;
}

// ---------------------------------------------------------------------------
// Kernel 7: spatial mean over 32x32, NHWC f16 -> f32 out[b][co]. grid=B, blk=128.
// ---------------------------------------------------------------------------
__global__ __launch_bounds__(128)
void mean_nhwc_kernel(const _Float16* __restrict__ h, float* __restrict__ out) {
  const int b  = blockIdx.x;
  const int co = threadIdx.x;                  // CO = 128
  const _Float16* p = h + (size_t)b * 1024 * 128 + co;
  float s = 0.f;
  for (int i = 0; i < 1024; ++i) s += (float)p[(size_t)i * 128];
  out[b * 128 + co] = s * (1.0f / 1024.0f);
}

// ---------------------------------------------------------------------------
extern "C" void kernel_launch(void* const* d_in, const int* in_sizes, int n_in,
                              void* d_out, int out_size, void* d_ws, size_t ws_size,
                              hipStream_t stream) {
  (void)in_sizes; (void)n_in; (void)out_size; (void)ws_size;
  const float* x   = (const float*)d_in[0];
  const float* w1  = (const float*)d_in[1];
  const float* b1  = (const float*)d_in[2];
  const float* g1  = (const float*)d_in[3];
  const float* bt1 = (const float*)d_in[4];
  const float* rm1 = (const float*)d_in[5];
  const float* rv1 = (const float*)d_in[6];
  const float* w2  = (const float*)d_in[7];
  const float* b2  = (const float*)d_in[8];
  const float* g2  = (const float*)d_in[9];
  const float* bt2 = (const float*)d_in[10];
  const float* rm2 = (const float*)d_in[11];
  const float* rv2 = (const float*)d_in[12];
  const float* w3  = (const float*)d_in[13];
  const float* b3  = (const float*)d_in[14];
  const float* g3  = (const float*)d_in[15];
  const float* bt3 = (const float*)d_in[16];
  const float* rm3 = (const float*)d_in[17];
  const float* rv3 = (const float*)d_in[18];
  float* out = (float*)d_out;

  // Workspace layout (f32 offsets, all 16B aligned). Peak ~51.5 MB.
  float* ws   = (float*)d_ws;
  float* tsum = ws;                                //  6144 f32
  float* tsq  = ws + 6144;                         //  6144 f32
  int*   idx  = (int*)(ws + 12288);                //    32 int
  _Float16* wpk1 = (_Float16*)(ws + 16384);        // [12][32][8]  =  3,072 f16
  _Float16* wpk2 = wpk1 + 3072;                    // [9][64][32]  = 18,432 f16
  _Float16* wpk3 = wpk2 + 18432;                   // [9][128][64] = 73,728 f16
  _Float16* sel  = (_Float16*)(ws + 64000);        // 32*131*130*8 =  4,359,680 f16
  _Float16* bufA = (_Float16*)(ws + 2243840);      // max 32*128*128*32 = 16,777,216 f16
  _Float16* bufB = (_Float16*)(ws + 10632448);     // max 32*66*66*32   =  4,460,544 f16

  const int B = 32;
  const int selN = 32 * 131 * 130 * 8;   // padded sel f16 count
  const int p1N  = 32 * 66 * 66 * 32;    // padded pool1-out f16 count
  const int p2N  = 32 * 34 * 34 * 64;    // padded pool2-out f16 count

  // Halo init (must run every call; harness does not re-poison between replays)
  fill_zero_kernel<<<(selN / 8 + 255) / 256, 256, 0, stream>>>(sel, selN / 8);

  // 1) tile stats (single pass over input) -> 2) variance argmin -> 3) gather
  tile_stats_kernel<<<B * 3 * 64, 256, 0, stream>>>(x, tsum, tsq);
  select_patch_kernel<<<B, 64, 0, stream>>>(tsum, tsq, idx);
  gather_patch_kernel<<<B * 128, 128, 0, stream>>>(x, idx, sel);

  // Weight repacks (tiny)
  repack_weights_kernel<<<(3072 + 255) / 256, 256, 0, stream>>>(w1, wpk1, 3, 8, 32, 3072);
  repack_weights_kernel<<<(18432 + 255) / 256, 256, 0, stream>>>(w2, wpk2, 32, 32, 64, 18432);
  repack_weights_kernel<<<(73728 + 255) / 256, 256, 0, stream>>>(w3, wpk3, 64, 64, 128, 73728);

  // 4) conv1: 8(pad of 3) -> 32, 128x128, PR=131, NT=2 (sel -> bufA). 4096 blocks
  conv_bn_relu_wmma_kernel<8, 32, 128, 128, 131, 2><<<4096, 256, 0, stream>>>(
      sel, wpk1, b1, g1, bt1, rm1, rv1, bufA);
  // 5) pool1: 32ch 128 -> 64, into padded 66x66 (bufA -> bufB)
  fill_zero_kernel<<<(p1N / 8 + 255) / 256, 256, 0, stream>>>(bufB, p1N / 8);
  {
    const int total = B * 32 * 64 * 64;
    maxpool2_nhwc_kernel<<<(total + 255) / 256, 256, 0, stream>>>(bufA, bufB, 32, 128, 128, total);
  }
  // 6) conv2: 32 -> 64, 64x64, PR=66, NT=4 (bufB -> bufA). 8192 waves -> 1024 blocks
  conv_bn_relu_wmma_kernel<32, 64, 64, 64, 66, 4><<<1024, 256, 0, stream>>>(
      bufB, wpk2, b2, g2, bt2, rm2, rv2, bufA);
  // 7) pool2: 64ch 64 -> 32, into padded 34x34 (bufA -> bufB)
  fill_zero_kernel<<<(p2N / 8 + 255) / 256, 256, 0, stream>>>(bufB, p2N / 8);
  {
    const int total = B * 64 * 32 * 32;
    maxpool2_nhwc_kernel<<<(total + 255) / 256, 256, 0, stream>>>(bufA, bufB, 64, 64, 64, total);
  }
  // 8) conv3: 64 -> 128, 32x32, PR=34, NT=4 (bufB -> bufA). 4096 waves -> 512 blocks
  conv_bn_relu_wmma_kernel<64, 128, 32, 32, 34, 4><<<512, 256, 0, stream>>>(
      bufB, wpk3, b3, g3, bt3, rm3, rv3, bufA);
  // 9) spatial mean -> out (32 x 128)
  mean_nhwc_kernel<<<B, 128, 0, stream>>>(bufA, out);
}
